// PositionalSelfAttention_4415226380579
// MI455X (gfx1250) — matrix operand order repaired and checked by reading, not compiled, not run on hardware
//
#include <hip/hip_runtime.h>
#include <hip/hip_fp16.h>
#include <math.h>

// ---------------------------------------------------------------------------
// Types
// ---------------------------------------------------------------------------
typedef __attribute__((ext_vector_type(16))) _Float16 v16h;
typedef __attribute__((ext_vector_type(8)))  float    v8f;
typedef __attribute__((ext_vector_type(4)))  float    f32x4;

union Frag {            // one WMMA 16-bit operand per lane (8 VGPRs)
  v16h  v;
  f32x4 f4[2];
};

// CDNA5 async global->LDS copy (16B per lane), tracked by ASYNCcnt.
// lds_addr: 32-bit LDS byte offset (flat-LDS aperture keeps offset in addr[31:0])
__device__ __forceinline__ void async_copy_b128(unsigned lds_addr, const void* gptr)
{
  asm volatile("global_load_async_to_lds_b128 %0, %1, off"
               :: "v"(lds_addr), "v"(gptr)
               : "memory");
}
__device__ __forceinline__ void wait_asynccnt_le4()
{
  asm volatile("s_wait_asynccnt 0x4" ::: "memory");
}
__device__ __forceinline__ void wait_asynccnt_0()
{
  asm volatile("s_wait_asynccnt 0x0" ::: "memory");
}
__device__ __forceinline__ unsigned lds_off(const void* p)
{
  return (unsigned)(size_t)p;    // LDS offset lives in addr[31:0]
}

// ---------------------------------------------------------------------------
// Tiled WMMA GEMM:  C[M,N] = A[M,K] * Brows[N,K]^T   (contraction on last dims)
// Block tile 128x128, 8 waves (4 along M x 2 along N), wave tile 32x64,
// K step 32 via v_wmma_f32_16x16x32_f16. Double-buffered LDS filled with
// global_load_async_to_lds_b128 one K-tile ahead (ASYNCcnt in-order).
//
// MODE 0: outF16[m*N+n] = acc + bias[n]                (row-major f16)
// MODE 1: outF16[b][n][s] = acc + bias[n]              (transposed f16, vT)
// MODE 2: outF32[m*N+n] = acc + bias[n] + res[m*N+n]   (f32 + residual)
// MODE 3: outF32[m*N+n] = acc*scale + posb[m*N+n], masked -> scores
// ---------------------------------------------------------------------------
#define BM 128
#define BN 128
#define BK 32
#define LDS_STRIDE 40   // halves; 80B row stride, 16B aligned

template <int MODE>
__global__ __launch_bounds__(256)
void gemm_wmma(const _Float16* __restrict__ A, long strideAb,
               const _Float16* __restrict__ Bm, long strideBb,
               float* __restrict__ outF32, _Float16* __restrict__ outF16,
               long strideCb,
               int N, int K, int lda, int ldb,
               const float* __restrict__ bias,
               const float* __restrict__ res,
               const float* __restrict__ posb,
               const int*  __restrict__ mask, long strideMaskB,
               float scoreScale, int ldT, int Sdim)
{
  const int b = blockIdx.z;
  A  += (long)b * strideAb;
  Bm += (long)b * strideBb;

  const int tid   = threadIdx.x;
  const int lane  = tid & 31;
  const int wave  = tid >> 5;
  const int waveM = wave & 3;     // 0..3
  const int waveN = wave >> 2;    // 0..1
  const int blockM = blockIdx.y * BM;
  const int blockN = blockIdx.x * BN;

  __shared__ __align__(16) _Float16 As[2][BM * LDS_STRIDE];
  __shared__ __align__(16) _Float16 Bs[2][BN * LDS_STRIDE];

  v8f acc[2][4] = {};

  // staging: 256 threads, each thread copies 2x16B of A and 2x16B of B per tile
  const int ldRow = tid >> 1;          // 0..127
  const int ldCol = (tid & 1) * 16;    // 0 or 16 (halves)

  const _Float16* gA = A  + (long)(blockM + ldRow) * lda + ldCol;
  const _Float16* gB = Bm + (long)(blockN + ldRow) * ldb + ldCol;
  const unsigned  lA0 = lds_off(&As[0][ldRow * LDS_STRIDE + ldCol]);
  const unsigned  lA1 = lds_off(&As[1][ldRow * LDS_STRIDE + ldCol]);
  const unsigned  lB0 = lds_off(&Bs[0][ldRow * LDS_STRIDE + ldCol]);
  const unsigned  lB1 = lds_off(&Bs[1][ldRow * LDS_STRIDE + ldCol]);

  const int r16 = lane & 15;
  const int hi  = lane >> 4;           // 0 or 1

  const int nk = K / BK;

  // prologue: async-stage tile 0 into buffer 0
  async_copy_b128(lA0,      gA);
  async_copy_b128(lA0 + 16, gA + 8);
  async_copy_b128(lB0,      gB);
  async_copy_b128(lB0 + 16, gB + 8);

  for (int i = 0; i < nk; ++i) {
    const int cur = i & 1;
    if (i + 1 < nk) {
      const _Float16* nA = gA + (long)(i + 1) * BK;
      const _Float16* nB = gB + (long)(i + 1) * BK;
      const unsigned  dA = cur ? lA0 : lA1;
      const unsigned  dB = cur ? lB0 : lB1;
      async_copy_b128(dA,      nA);
      async_copy_b128(dA + 16, nA + 8);
      async_copy_b128(dB,      nB);
      async_copy_b128(dB + 16, nB + 8);
      wait_asynccnt_le4();    // the 4 older ops (tile i) are complete
    } else {
      wait_asynccnt_0();
    }
    __syncthreads();          // tile i visible to all waves

    Frag afr[2], bfr[4];
#pragma unroll
    for (int mt = 0; mt < 2; ++mt) {
      // A layout: lane<16 -> K {0..7,16..23}; lane>=16 -> K {8..15,24..31}
      const _Float16* p = &As[cur][(waveM * 32 + mt * 16 + r16) * LDS_STRIDE + hi * 8];
      afr[mt].f4[0] = *reinterpret_cast<const f32x4*>(p);
      afr[mt].f4[1] = *reinterpret_cast<const f32x4*>(p + 16);
    }
#pragma unroll
    for (int nt = 0; nt < 4; ++nt) {
      // B layout: lane<16 -> K {0..7,16..23}; lane>=16 -> K {8..15,24..31}
      const _Float16* p = &Bs[cur][(waveN * 64 + nt * 16 + r16) * LDS_STRIDE + hi * 8];
      bfr[nt].f4[0] = *reinterpret_cast<const f32x4*>(p);
      bfr[nt].f4[1] = *reinterpret_cast<const f32x4*>(p + 16);
    }
#pragma unroll
    for (int mt = 0; mt < 2; ++mt)
#pragma unroll
      for (int nt = 0; nt < 4; ++nt)
        acc[mt][nt] = __builtin_amdgcn_wmma_f32_16x16x32_f16(
            false, afr[mt].v, false, bfr[nt].v, (short)0, acc[mt][nt],
            false, false);

    __syncthreads();          // reads of buf cur done before it is refilled
  }

  // Epilogue. D layout: lane -> n = lane&15, m = 8*(lane>>4) + vgpr_index
  const int mOff = hi * 8;
#pragma unroll
  for (int mt = 0; mt < 2; ++mt) {
#pragma unroll
    for (int nt = 0; nt < 4; ++nt) {
      const int gn  = blockN + waveN * 64 + nt * 16 + r16;
      const int gm0 = blockM + waveM * 32 + mt * 16 + mOff;
      if (MODE == 0) {
        const float bvv = bias ? bias[gn] : 0.0f;
        _Float16* o = outF16 + (long)b * strideCb;
#pragma unroll
        for (int r = 0; r < 8; ++r)
          o[(long)(gm0 + r) * N + gn] = (_Float16)(acc[mt][nt][r] + bvv);
      } else if (MODE == 1) {
        const float bvv = bias ? bias[gn] : 0.0f;
        const int bb = gm0 / Sdim;
        const int s0 = gm0 % Sdim;          // 8-aligned, one batch per tile
        union { f32x4 f4; _Float16 h[8]; } pack;
#pragma unroll
        for (int r = 0; r < 8; ++r) pack.h[r] = (_Float16)(acc[mt][nt][r] + bvv);
        *reinterpret_cast<f32x4*>(outF16 + ((long)bb * N + gn) * ldT + s0) = pack.f4;
      } else if (MODE == 2) {
        const float bvv = bias[gn];
#pragma unroll
        for (int r = 0; r < 8; ++r) {
          const long idx = (long)(gm0 + r) * N + gn;
          outF32[idx] = acc[mt][nt][r] + bvv + res[idx];
        }
      } else { // MODE 3: scores
        float* o = outF32 + (long)b * strideCb;
        const int* mk = mask + (long)b * strideMaskB;
#pragma unroll
        for (int r = 0; r < 8; ++r) {
          const long idx = (long)(gm0 + r) * N + gn;
          float val = acc[mt][nt][r] * scoreScale + posb[idx];
          if (mk[idx] == 0) val = -1e9f;
          o[idx] = val;
        }
      }
    }
  }
}

// ---------------------------------------------------------------------------
// xp = x + sinusoidal PE, converted to f16
// ---------------------------------------------------------------------------
__global__ void add_pe_f16(const float* __restrict__ x, _Float16* __restrict__ xp,
                           int S, int H, long total)
{
  long i = (long)blockIdx.x * blockDim.x + threadIdx.x;
  if (i >= total) return;
  const int hcol = (int)(i % H);
  const int s    = (int)((i / H) % S);
  const int twoI = hcol & ~1;
  const float ang = (float)s * __expf((float)twoI * (-logf(10000.0f) / (float)H));
  const float pe  = (hcol & 1) ? __cosf(ang) : __sinf(ang);
  xp[i] = (_Float16)(x[i] + pe);
}

__global__ void f32_to_f16(const float* __restrict__ in, _Float16* __restrict__ out, long n)
{
  long i = (long)blockIdx.x * blockDim.x + threadIdx.x;
  if (i < n) out[i] = (_Float16)in[i];
}

// ---------------------------------------------------------------------------
// Row softmax over S=2048 (one block / row): in-place f32 + f16 copy
// ---------------------------------------------------------------------------
__global__ __launch_bounds__(256)
void softmax_rows(float* __restrict__ attn, _Float16* __restrict__ attn16)
{
  const long row = blockIdx.x;
  float*     p   = attn   + row * 2048L;
  _Float16*  p16 = attn16 + row * 2048L;
  const int tid = threadIdx.x;
  __shared__ float red[256];

  float vals[8];
  float m = -3.402823466e38f;
#pragma unroll
  for (int i = 0; i < 8; ++i) { vals[i] = p[tid + i * 256]; m = fmaxf(m, vals[i]); }
  red[tid] = m; __syncthreads();
  for (int s = 128; s > 0; s >>= 1) {
    if (tid < s) red[tid] = fmaxf(red[tid], red[tid + s]);
    __syncthreads();
  }
  m = red[0]; __syncthreads();

  float sum = 0.0f;
#pragma unroll
  for (int i = 0; i < 8; ++i) { vals[i] = __expf(vals[i] - m); sum += vals[i]; }
  red[tid] = sum; __syncthreads();
  for (int s = 128; s > 0; s >>= 1) {
    if (tid < s) red[tid] += red[tid + s];
    __syncthreads();
  }
  const float inv = 1.0f / red[0];
#pragma unroll
  for (int i = 0; i < 8; ++i) {
    const float a = vals[i] * inv;
    p[tid + i * 256]   = a;
    p16[tid + i * 256] = (_Float16)a;
  }
}

// ---------------------------------------------------------------------------
// LayerNorm over H=1024 (one block / row)
// ---------------------------------------------------------------------------
__global__ __launch_bounds__(256)
void layernorm_rows(const float* __restrict__ h, const float* __restrict__ gamma,
                    const float* __restrict__ beta, float* __restrict__ out)
{
  const long row = blockIdx.x;
  const float* p = h + row * 1024L;
  float*       o = out + row * 1024L;
  const int tid = threadIdx.x;
  __shared__ float r1[256], r2[256];

  float v[4];
  float s = 0.0f, ss = 0.0f;
#pragma unroll
  for (int i = 0; i < 4; ++i) {
    v[i] = p[tid + i * 256];
    s  += v[i];
    ss += v[i] * v[i];
  }
  r1[tid] = s; r2[tid] = ss; __syncthreads();
  for (int st = 128; st > 0; st >>= 1) {
    if (tid < st) { r1[tid] += r1[tid + st]; r2[tid] += r2[tid + st]; }
    __syncthreads();
  }
  const float mu  = r1[0] * (1.0f / 1024.0f);
  const float var = r2[0] * (1.0f / 1024.0f) - mu * mu;
  const float inv = rsqrtf(var + 1e-5f);
#pragma unroll
  for (int i = 0; i < 4; ++i) {
    const int c = tid + i * 256;
    o[c] = (v[i] - mu) * inv * gamma[c] + beta[c];
  }
}

// ---------------------------------------------------------------------------
// Launcher
// ---------------------------------------------------------------------------
extern "C" void kernel_launch(void* const* d_in, const int* in_sizes, int n_in,
                              void* d_out, int out_size, void* d_ws, size_t ws_size,
                              hipStream_t stream)
{
  (void)in_sizes; (void)n_in; (void)out_size; (void)ws_size;
  const int B = 4, S = 2048, H = 1024;

  const float* x     = (const float*)d_in[0];
  const int*   mask  = (const int*)  d_in[1];
  const float* Wq    = (const float*)d_in[2];
  const float* bq    = (const float*)d_in[3];
  const float* Wk    = (const float*)d_in[4];
  const float* bk    = (const float*)d_in[5];
  const float* Wv    = (const float*)d_in[6];
  const float* bv    = (const float*)d_in[7];
  const float* posb  = (const float*)d_in[8];
  const float* Wo    = (const float*)d_in[9];
  const float* bo    = (const float*)d_in[10];
  const float* gamma = (const float*)d_in[11];
  const float* beta  = (const float*)d_in[12];

  float* out_norm = (float*)d_out;
  float* out_attn = out_norm + (long)B * S * H;

  // workspace layout (f16 buffers), with reuse:
  char* ws = (char*)d_ws;
  _Float16* xp16 = (_Float16*)ws; ws += (long)B * S * H * 2;   // 16.8 MB
  _Float16* Wq16 = (_Float16*)ws; ws += (long)H * H * 2;       //  2.1 MB
  _Float16* Wk16 = (_Float16*)ws; ws += (long)H * H * 2;
  _Float16* Wv16 = (_Float16*)ws; ws += (long)H * H * 2;
  _Float16* Wo16 = (_Float16*)ws; ws += (long)H * H * 2;
  _Float16* q16  = (_Float16*)ws; ws += (long)B * S * H * 2;   // 16.8 MB
  _Float16* k16  = (_Float16*)ws; ws += (long)B * S * H * 2;   // 16.8 MB
  _Float16* vT16 = (_Float16*)ws; ws += (long)B * S * H * 2;   // 16.8 MB [B][H][S]
  _Float16* a16  = (_Float16*)ws; ws += (long)B * S * S * 2;   // 33.6 MB
  _Float16* ctx16 = q16;                 // reuse: q dead after scores
  float*    hbuf  = (float*)k16;         // reuse: k16+vT16 = B*S*H*4 bytes exactly

  const dim3 blk(256);

  // 1) xp = x + PE  (f16)
  {
    long tot = (long)B * S * H;
    add_pe_f16<<<(unsigned)((tot + 255) / 256), blk, 0, stream>>>(x, xp16, S, H, tot);
  }
  // 2) weight conversions
  {
    long hh = (long)H * H;
    unsigned g = (unsigned)((hh + 255) / 256);
    f32_to_f16<<<g, blk, 0, stream>>>(Wq, Wq16, hh);
    f32_to_f16<<<g, blk, 0, stream>>>(Wk, Wk16, hh);
    f32_to_f16<<<g, blk, 0, stream>>>(Wv, Wv16, hh);
    f32_to_f16<<<g, blk, 0, stream>>>(Wo, Wo16, hh);
  }
  // 3) Q, K projections (row-major f16) and V projection (transposed store)
  {
    dim3 g1(H / BN, (B * S) / BM, 1);
    gemm_wmma<0><<<g1, blk, 0, stream>>>(xp16, 0, Wq16, 0, nullptr, q16, 0,
        H, H, H, H, bq, nullptr, nullptr, nullptr, 0, 0.0f, 0, S);
    gemm_wmma<0><<<g1, blk, 0, stream>>>(xp16, 0, Wk16, 0, nullptr, k16, 0,
        H, H, H, H, bk, nullptr, nullptr, nullptr, 0, 0.0f, 0, S);
    gemm_wmma<1><<<g1, blk, 0, stream>>>(xp16, 0, Wv16, 0, nullptr, vT16, 0,
        H, H, H, H, bv, nullptr, nullptr, nullptr, 0, 0.0f, S, S);
  }
  // 4) scores = q k^T / sqrt(H) + pos_bias, masked  -> d_out attn region (f32)
  {
    dim3 g2(S / BN, S / BM, B);
    gemm_wmma<3><<<g2, blk, 0, stream>>>(q16, (long)S * H, k16, (long)S * H,
        out_attn, nullptr, (long)S * S,
        S, H, H, H, nullptr, nullptr, posb, mask, (long)S * S,
        1.0f / sqrtf((float)H), 0, S);
  }
  // 5) softmax rows (in place) + f16 copy
  softmax_rows<<<B * S, blk, 0, stream>>>(out_attn, a16);

  // 6) context = attn @ v   (A = attn16 rows, B = vT rows)
  {
    dim3 g3(H / BN, S / BM, B);
    gemm_wmma<0><<<g3, blk, 0, stream>>>(a16, (long)S * S, vT16, (long)H * S,
        nullptr, ctx16, (long)S * H,
        H, S, S, S, nullptr, nullptr, nullptr, nullptr, 0, 0.0f, 0, S);
  }
  // 7) out = ctx @ Wo^T + bo + x  (f32 pre-LN buffer)
  {
    dim3 g4(H / BN, (B * S) / BM, 1);
    gemm_wmma<2><<<g4, blk, 0, stream>>>(ctx16, 0, Wo16, 0, hbuf, nullptr, 0,
        H, H, H, H, bo, x, nullptr, nullptr, 0, 0.0f, 0, S);
  }
  // 8) LayerNorm -> d_out normed region
  layernorm_rows<<<B * S, blk, 0, stream>>>(hbuf, gamma, beta, out_norm);
}
